// Mamba_2508260901097
// MI455X (gfx1250) — compile-verified, matrix operand order checked by reading, not run on hardware
//
#include <hip/hip_runtime.h>

typedef __bf16 bf16;
typedef __attribute__((ext_vector_type(16))) __bf16 v16bf;
typedef __attribute__((ext_vector_type(8)))  __bf16 bf16x8;
typedef __attribute__((ext_vector_type(8)))  float  v8f;
typedef unsigned int u32x4 __attribute__((ext_vector_type(4)));
typedef int          i32x4 __attribute__((ext_vector_type(4)));
typedef int          i32x8 __attribute__((ext_vector_type(8)));

#define DMODEL 1024
#define DIN    2048
#define DSTATE 16
#define DTRK   64
#define NLAYER 4
#define BB     2
#define LL     2048
#define MM     (BB*LL)
#define CHUNK_L 64
#define NCHUNK  (LL/CHUNK_L)

#if __has_builtin(__builtin_amdgcn_tensor_load_to_lds) && __has_builtin(__builtin_amdgcn_s_wait_tensorcnt)
#define USE_TDM 1
#else
#define USE_TDM 0
#endif

// ----------------------------- conversion kernels ---------------------------
__global__ void cvt_f32_bf16_kernel(const float* __restrict__ src,
                                    bf16* __restrict__ dst, int n) {
  int i = blockIdx.x * 256 + threadIdx.x;
  if (i < n) dst[i] = (bf16)src[i];
}

// extract dt-rank columns (first 64 of 96) of x_dbl as a packed bf16 matrix
__global__ void cvt_dtr_kernel(const float* __restrict__ xdbl,
                               bf16* __restrict__ dtr, int m) {
  int i = blockIdx.x * 256 + threadIdx.x;
  if (i < m * DTRK) {
    int r = i >> 6, k = i & 63;
    dtr[i] = (bf16)xdbl[(size_t)r * 96 + k];
  }
}

// ----------------------------- layernorm ------------------------------------
__global__ __launch_bounds__(256) void layernorm_kernel(
    const float* __restrict__ initsrc, float* __restrict__ resid,
    const float* __restrict__ w, const float* __restrict__ b,
    bf16* __restrict__ out_bf, float* __restrict__ out_f)
{
  int row = blockIdx.x;
  int t = threadIdx.x;
  size_t base = (size_t)row * DMODEL;
  float v[4]; float s = 0.f, s2 = 0.f;
#pragma unroll
  for (int i = 0; i < 4; i++) {
    int c = t + i * 256;
    float x;
    if (initsrc) { x = initsrc[base + c]; resid[base + c] = x; }
    else         { x = resid[base + c]; }
    v[i] = x; s += x; s2 += x * x;
  }
#pragma unroll
  for (int o = 16; o > 0; o >>= 1) {
    s  += __shfl_down(s,  o, 32);
    s2 += __shfl_down(s2, o, 32);
  }
  __shared__ float red0[8], red1[8];
  if ((t & 31) == 0) { red0[t >> 5] = s; red1[t >> 5] = s2; }
  __syncthreads();
  float ts = 0.f, ts2 = 0.f;
#pragma unroll
  for (int i = 0; i < 8; i++) { ts += red0[i]; ts2 += red1[i]; }
  float mu  = ts  * (1.f / DMODEL);
  float var = ts2 * (1.f / DMODEL) - mu * mu;
  float rs  = rsqrtf(var + 1e-5f);
#pragma unroll
  for (int i = 0; i < 4; i++) {
    int c = t + i * 256;
    float y = (v[i] - mu) * rs * w[c] + b[c];
    if (out_bf) out_bf[base + c] = (bf16)y;
    if (out_f)  out_f[base + c]  = y;
  }
}

// ----------------------------- WMMA GEMM ------------------------------------
// C[M,N] = A[M,K] * Bw[N,K]^T ; A,Bw bf16 row-major; f32 accumulate.
// 256 threads = 8 waves; block tile 128x64; wave tile 32x32 (2x2 WMMA tiles);
// K staged 64 at a time (two 16x16x32 WMMA k-steps per stage).
#define BM 128
#define BN 64
#define BK 64
#define LDT 72   // 64 + 8 pad bf16 elems -> 144B rows, keeps 16B alignment

enum { OUT_BF16 = 0, OUT_F32 = 1, OUT_F32_ADD = 2, OUT_SOFTPLUS = 3 };

// 16x32 bf16 fragment from LDS, per ISA A-layout:
// lanes 0-15:  row=lane,    K {0..7, 16..23}
// lanes 16-31: row=lane-16, K {8..15, 24..31}
__device__ inline v16bf frag_from_lds(const bf16* rowbase, int lane) {
  int r  = lane & 15;
  int kb = (lane < 16) ? 0 : 8;
  const bf16* p = rowbase + r * LDT + kb;
  bf16x8 lo = *(const bf16x8*)(p);
  bf16x8 hi = *(const bf16x8*)(p + 16);
  v16bf f;
#pragma unroll
  for (int i = 0; i < 8; i++) { f[i] = lo[i]; f[i + 8] = hi[i]; }
  return f;
}

__device__ inline void wmma_stage(const bf16* Ab, const bf16* Bb,
                                  int wm, int wn, int lane, v8f acc[2][2]) {
#pragma unroll
  for (int kk = 0; kk < BK; kk += 32) {
    v16bf a0 = frag_from_lds(Ab + (wm     ) * LDT + kk, lane);
    v16bf a1 = frag_from_lds(Ab + (wm + 16) * LDT + kk, lane);
    v16bf b0 = frag_from_lds(Bb + (wn     ) * LDT + kk, lane);
    v16bf b1 = frag_from_lds(Bb + (wn + 16) * LDT + kk, lane);
    acc[0][0] = __builtin_amdgcn_wmma_f32_16x16x32_bf16(false, a0, false, b0, (short)0, acc[0][0], false, false);
    acc[0][1] = __builtin_amdgcn_wmma_f32_16x16x32_bf16(false, a0, false, b1, (short)0, acc[0][1], false, false);
    acc[1][0] = __builtin_amdgcn_wmma_f32_16x16x32_bf16(false, a1, false, b0, (short)0, acc[1][0], false, false);
    acc[1][1] = __builtin_amdgcn_wmma_f32_16x16x32_bf16(false, a1, false, b1, (short)0, acc[1][1], false, false);
  }
}

#if USE_TDM
// Issue a TDM 2D tile load: tile (BK cols x tile_rows) of bf16 from a
// row-major tensor (row stride = stride elems), into LDS at lds_off with
// hardware padding: +4 DWORDs (8 bf16) after every 32 DWORDs (64 bf16) = LDT.
__device__ inline void tdm_load_tile(unsigned lds_off, const bf16* gptr,
                                     int tile_rows, int tens_rows,
                                     int tens_cols, int stride) {
  unsigned long long ga = (unsigned long long)(size_t)(const void*)gptr;
  u32x4 g0;
  g0[0] = 1u;                                             // count=1, user D#
  g0[1] = lds_off;                                        // LDS byte address
  g0[2] = (unsigned)(ga & 0xFFFFFFFFull);                 // global addr lo
  g0[3] = (unsigned)((ga >> 32) & 0x1FFFFFFull) | (2u << 30);  // hi + type=2
  unsigned td0 = (unsigned)tens_cols, td1 = (unsigned)tens_rows;
  unsigned long long st = (unsigned long long)stride;
  i32x8 g1;
  g1[0] = (int)((1u << 16) |            // data_size = 2 bytes
                (1u << 20) |            // pad_enable
                (4u << 22) |            // pad_interval: 32 DWORDs
                (3u << 25));            // pad_amount:    4 DWORDs
  g1[1] = (int)((td0 & 0xFFFFu) << 16);                   // tensor_dim0 lo
  g1[2] = (int)(((td0 >> 16) & 0xFFFFu) | ((td1 & 0xFFFFu) << 16));
  g1[3] = (int)(((td1 >> 16) & 0xFFFFu) | ((unsigned)BK << 16)); // tile_dim0
  g1[4] = (int)((unsigned)tile_rows & 0xFFFFu);           // tile_dim1, dim2=0
  g1[5] = (int)(st & 0xFFFFFFFFull);                      // dim0 stride lo
  g1[6] = (int)((st >> 32) & 0xFFFFull);                  // stride hi, d1s=0
  g1[7] = 0;
  i32x4 z = {0, 0, 0, 0};
#if __has_include(<hip/amd_detail/amd_gfx1250_TDM.h>)
  i32x8 z8 = {0, 0, 0, 0, 0, 0, 0, 0};
  __builtin_amdgcn_tensor_load_to_lds(g0, g1, z, z, z8, 0);
#else
  __builtin_amdgcn_tensor_load_to_lds(g0, g1, z, z, 0);
#endif
}
#endif

__global__ __launch_bounds__(256) void wmma_gemm_kernel(
    const bf16* __restrict__ A, const bf16* __restrict__ Bw,
    void* __restrict__ outp, const float* __restrict__ aux,
    int M, int N, int K, int mode)
{
  __shared__ __align__(16) bf16 As[2][BM * LDT];
  __shared__ __align__(16) bf16 Bs[2][BN * LDT];
  const int tid  = threadIdx.x;
  const int wave = tid >> 5, lane = tid & 31;
  const int m0 = blockIdx.y * BM;
  const int n0 = blockIdx.x * BN;
  const int wm = (wave >> 1) * 32;   // 0,32,64,96
  const int wn = (wave & 1) * 32;    // 0,32
  v8f acc[2][2] = {};
  const int nt = K / BK;

#if USE_TDM
  // wave 0 DMAs A tiles, wave 1 DMAs B tiles; double-buffered over LDS.
  if (wave == 0) {
    tdm_load_tile((unsigned)(size_t)(void*)&As[0][0], A + (size_t)m0 * K,
                  BM, M - m0, K, K);
    if (nt > 1)
      tdm_load_tile((unsigned)(size_t)(void*)&As[1][0], A + (size_t)m0 * K + BK,
                    BM, M - m0, K - BK, K);
  }
  if (wave == 1) {
    tdm_load_tile((unsigned)(size_t)(void*)&Bs[0][0], Bw + (size_t)n0 * K,
                  BN, N - n0, K, K);
    if (nt > 1)
      tdm_load_tile((unsigned)(size_t)(void*)&Bs[1][0], Bw + (size_t)n0 * K + BK,
                    BN, N - n0, K - BK, K);
  }
  for (int t = 0; t < nt; ++t) {
    if (wave < 2) {
      if (t + 1 < nt) __builtin_amdgcn_s_wait_tensorcnt((short)1);
      else            __builtin_amdgcn_s_wait_tensorcnt((short)0);
    }
    __syncthreads();                     // tile t resident for all waves
    wmma_stage(&As[t & 1][0], &Bs[t & 1][0], wm, wn, lane, acc);
    __syncthreads();                     // all waves done reading buffer t&1
    if (t + 2 < nt) {
      int k2 = (t + 2) * BK;
      if (wave == 0)
        tdm_load_tile((unsigned)(size_t)(void*)&As[t & 1][0],
                      A + (size_t)m0 * K + k2, BM, M - m0, K - k2, K);
      if (wave == 1)
        tdm_load_tile((unsigned)(size_t)(void*)&Bs[t & 1][0],
                      Bw + (size_t)n0 * K + k2, BN, N - n0, K - k2, K);
    }
  }
#else
  // fallback: cooperative VALU staging, single buffer
  for (int t = 0; t < nt; ++t) {
    int k0 = t * BK;
    {   // A tile 128x64: each thread 64B
      int r = tid >> 1, c = (tid & 1) * 32;
      int gr = m0 + r; if (gr >= M) gr = M - 1;
      const bf16* src = A + (size_t)gr * K + k0 + c;
      if (k0 + BK < K) __builtin_prefetch(src + BK, 0, 3);
#pragma unroll
      for (int q = 0; q < 4; q++)
        *(bf16x8*)&As[0][r * LDT + c + q * 8] = *(const bf16x8*)(src + q * 8);
    }
    {   // B tile 64x64: each thread 32B
      int r = tid >> 2, c = (tid & 3) * 16;
      int gn = n0 + r; if (gn >= N) gn = N - 1;
      const bf16* src = Bw + (size_t)gn * K + k0 + c;
      if (k0 + BK < K) __builtin_prefetch(src + BK, 0, 3);
#pragma unroll
      for (int q = 0; q < 2; q++)
        *(bf16x8*)&Bs[0][r * LDT + c + q * 8] = *(const bf16x8*)(src + q * 8);
    }
    __syncthreads();
    wmma_stage(&As[0][0], &Bs[0][0], wm, wn, lane, acc);
    __syncthreads();
  }
#endif

  // D layout: VGPR r -> M = r (+8 for lanes 16-31), N = lane%16
  const int mrow = (lane < 16) ? 0 : 8;
  const int ncol = lane & 15;
#pragma unroll
  for (int i = 0; i < 2; i++)
#pragma unroll
    for (int j = 0; j < 2; j++)
#pragma unroll
      for (int r = 0; r < 8; r++) {
        int gm = m0 + wm + i * 16 + r + mrow;
        int gn = n0 + wn + j * 16 + ncol;
        if (gm < M && gn < N) {
          float vv = acc[i][j][r];
          size_t idx = (size_t)gm * N + gn;
          if (mode == OUT_BF16)         ((bf16*)outp)[idx] = (bf16)vv;
          else if (mode == OUT_F32)     ((float*)outp)[idx] = vv;
          else if (mode == OUT_F32_ADD) ((float*)outp)[idx] += vv;  // residual
          else {                                                    // softplus
            float tt = vv + aux[gn];
            ((float*)outp)[idx] = (tt > 20.f) ? tt : log1pf(__expf(tt));
          }
        }
      }
}

// ----------------------------- depthwise conv + SiLU ------------------------
__global__ void conv_silu_kernel(const bf16* __restrict__ xz,
                                 const float* __restrict__ cw,
                                 const float* __restrict__ cb,
                                 bf16* __restrict__ xc)
{
  int i = blockIdx.x * 256 + threadIdx.x;
  if (i >= MM * DIN) return;
  int d  = i & (DIN - 1);
  int bl = i / DIN;
  int l  = bl & (LL - 1);
  int b  = bl / LL;
  float acc = cb[d];
#pragma unroll
  for (int j = 0; j < 4; j++) {
    int ls = l - 3 + j;
    if (ls >= 0) acc += cw[d * 4 + j] * (float)xz[((size_t)b * LL + ls) * (2 * DIN) + d];
  }
  float sv = acc / (1.f + __expf(-acc));
  xc[i] = (bf16)sv;
}

// ----------------------------- chunked selective scan -----------------------
__global__ __launch_bounds__(256) void scan_partial_kernel(
    const float* __restrict__ dt, const bf16* __restrict__ xc,
    const float* __restrict__ xdbl, const float* __restrict__ A_log,
    float* __restrict__ Aprod, float* __restrict__ Hpart)
{
  const int c  = blockIdx.y;
  const int bd = blockIdx.x * 256 + threadIdx.x;   // b*DIN + d
  const int b = bd / DIN, d = bd % DIN;
  __shared__ float Bsh[CHUNK_L][DSTATE];
  for (int i = threadIdx.x; i < CHUNK_L * DSTATE; i += 256) {
    int l = i >> 4, n = i & 15;
    Bsh[l][n] = xdbl[((size_t)b * LL + c * CHUNK_L + l) * 96 + DTRK + n];
  }
  __syncthreads();
  float Av[DSTATE], ap[DSTATE], hp[DSTATE];
#pragma unroll
  for (int n = 0; n < DSTATE; n++) {
    Av[n] = -__expf(A_log[d * DSTATE + n]);
    ap[n] = 1.f; hp[n] = 0.f;
  }
  for (int l = 0; l < CHUNK_L; l++) {
    size_t li = ((size_t)b * LL + c * CHUNK_L + l) * DIN + d;
    float dtv = dt[li];
    float dx  = dtv * (float)xc[li];
#pragma unroll
    for (int n = 0; n < DSTATE; n++) {
      float a = __expf(dtv * Av[n]);
      hp[n] = a * hp[n] + dx * Bsh[l][n];
      ap[n] *= a;
    }
  }
  size_t o = (((size_t)c * BB + b) * DIN + d) * DSTATE;
#pragma unroll
  for (int n = 0; n < DSTATE; n++) { Aprod[o + n] = ap[n]; Hpart[o + n] = hp[n]; }
}

__global__ void scan_carry_kernel(const float* __restrict__ Aprod,
                                  const float* __restrict__ Hpart,
                                  float* __restrict__ Hinit)
{
  int i = blockIdx.x * 256 + threadIdx.x;
  const size_t stride = (size_t)BB * DIN * DSTATE;
  if (i >= (int)stride) return;
  float h = 0.f;
  for (int c = 0; c < NCHUNK; c++) {
    Hinit[c * stride + i] = h;
    h = Aprod[c * stride + i] * h + Hpart[c * stride + i];
  }
}

__global__ __launch_bounds__(256) void scan_final_kernel(
    const float* __restrict__ dt, const bf16* __restrict__ xc,
    const float* __restrict__ xdbl, const bf16* __restrict__ xz,
    const float* __restrict__ A_log, const float* __restrict__ Dp,
    const float* __restrict__ Hinit, bf16* __restrict__ y)
{
  const int c  = blockIdx.y;
  const int bd = blockIdx.x * 256 + threadIdx.x;
  const int b = bd / DIN, d = bd % DIN;
  __shared__ float Bsh[CHUNK_L][DSTATE];
  __shared__ float Csh[CHUNK_L][DSTATE];
  for (int i = threadIdx.x; i < CHUNK_L * DSTATE; i += 256) {
    int l = i >> 4, n = i & 15;
    size_t rb = ((size_t)b * LL + c * CHUNK_L + l) * 96 + DTRK;
    Bsh[l][n] = xdbl[rb + n];
    Csh[l][n] = xdbl[rb + DSTATE + n];
  }
  __syncthreads();
  float Av[DSTATE], h[DSTATE];
  size_t o = (((size_t)c * BB + b) * DIN + d) * DSTATE;
#pragma unroll
  for (int n = 0; n < DSTATE; n++) {
    Av[n] = -__expf(A_log[d * DSTATE + n]);
    h[n]  = Hinit[o + n];
  }
  const float Dv = Dp[d];
  for (int l = 0; l < CHUNK_L; l++) {
    size_t li = ((size_t)b * LL + c * CHUNK_L + l) * DIN + d;
    float dtv = dt[li];
    float xv  = (float)xc[li];
    float dx  = dtv * xv;
    float yv  = 0.f;
#pragma unroll
    for (int n = 0; n < DSTATE; n++) {
      float a = __expf(dtv * Av[n]);
      h[n] = a * h[n] + dx * Bsh[l][n];
      yv  += h[n] * Csh[l][n];
    }
    yv += Dv * xv;
    float zv = (float)xz[((size_t)b * LL + c * CHUNK_L + l) * (2 * DIN) + DIN + d];
    yv *= zv / (1.f + __expf(-zv));
    y[li] = (bf16)yv;
  }
}

// ----------------------------- host orchestration ---------------------------
extern "C" void kernel_launch(void* const* d_in, const int* in_sizes, int n_in,
                              void* d_out, int out_size, void* d_ws, size_t ws_size,
                              hipStream_t stream)
{
  (void)in_sizes; (void)n_in; (void)out_size; (void)ws_size;
  const float* hs_in  = (const float*)d_in[0];
  const float* norm_w = (const float*)d_in[1];
  const float* norm_b = (const float*)d_in[2];
  const float* in_w   = (const float*)d_in[3];
  const float* conv_w = (const float*)d_in[4];
  const float* conv_b = (const float*)d_in[5];
  const float* xp_w   = (const float*)d_in[6];
  const float* dtp_w  = (const float*)d_in[7];
  const float* dtp_b  = (const float*)d_in[8];
  const float* A_log  = (const float*)d_in[9];
  const float* D_par  = (const float*)d_in[10];
  const float* out_w  = (const float*)d_in[11];
  const float* fn_w   = (const float*)d_in[12];
  const float* fn_b   = (const float*)d_in[13];

  size_t off = 0;
  auto carve = [&](size_t bytes) -> void* {
    void* p = (char*)d_ws + off;
    off += (bytes + 255) & ~(size_t)255;
    return p;
  };
  float* resid = (float*)carve((size_t)MM * DMODEL * 4);
  bf16*  hsn   = (bf16*) carve((size_t)MM * DMODEL * 2);
  bf16*  xzb   = (bf16*) carve((size_t)MM * 2 * DIN * 2);
  bf16*  xcb   = (bf16*) carve((size_t)MM * DIN * 2);
  float* xdbl  = (float*)carve((size_t)MM * 96 * 4);
  bf16*  dtr   = (bf16*) carve((size_t)MM * DTRK * 2);
  float* dtf   = (float*)carve((size_t)MM * DIN * 4);
  bf16*  yb    = (bf16*) carve((size_t)MM * DIN * 2);
  const size_t cst = (size_t)NCHUNK * BB * DIN * DSTATE;
  float* Aprod = (float*)carve(cst * 4);
  float* Hpart = (float*)carve(cst * 4);
  float* Hinit = (float*)carve(cst * 4);
  bf16* w_in  = (bf16*)carve((size_t)2 * DIN * DMODEL * 2);
  bf16* w_xp  = (bf16*)carve((size_t)96 * DIN * 2);
  bf16* w_dt  = (bf16*)carve((size_t)DIN * DTRK * 2);
  bf16* w_out = (bf16*)carve((size_t)DMODEL * DIN * 2);

  const dim3 blk(256);
  for (int l = 0; l < NLAYER; l++) {
    { int n = 2 * DIN * DMODEL; cvt_f32_bf16_kernel<<<(n + 255) / 256, blk, 0, stream>>>(in_w  + (size_t)l * n, w_in,  n); }
    { int n = 96 * DIN;         cvt_f32_bf16_kernel<<<(n + 255) / 256, blk, 0, stream>>>(xp_w  + (size_t)l * n, w_xp,  n); }
    { int n = DIN * DTRK;       cvt_f32_bf16_kernel<<<(n + 255) / 256, blk, 0, stream>>>(dtp_w + (size_t)l * n, w_dt,  n); }
    { int n = DMODEL * DIN;     cvt_f32_bf16_kernel<<<(n + 255) / 256, blk, 0, stream>>>(out_w + (size_t)l * n, w_out, n); }

    layernorm_kernel<<<MM, blk, 0, stream>>>(
        l == 0 ? hs_in : (const float*)nullptr, resid,
        norm_w + (size_t)l * DMODEL, norm_b + (size_t)l * DMODEL,
        hsn, (float*)nullptr);

    wmma_gemm_kernel<<<dim3((2 * DIN + BN - 1) / BN, MM / BM), blk, 0, stream>>>(
        hsn, w_in, xzb, (const float*)nullptr, MM, 2 * DIN, DMODEL, OUT_BF16);

    conv_silu_kernel<<<(MM * DIN + 255) / 256, blk, 0, stream>>>(
        xzb, conv_w + (size_t)l * DIN * 4, conv_b + (size_t)l * DIN, xcb);

    wmma_gemm_kernel<<<dim3((96 + BN - 1) / BN, MM / BM), blk, 0, stream>>>(
        xcb, w_xp, xdbl, (const float*)nullptr, MM, 96, DIN, OUT_F32);

    cvt_dtr_kernel<<<(MM * DTRK + 255) / 256, blk, 0, stream>>>(xdbl, dtr, MM);

    wmma_gemm_kernel<<<dim3((DIN + BN - 1) / BN, MM / BM), blk, 0, stream>>>(
        dtr, w_dt, dtf, dtp_b + (size_t)l * DIN, MM, DIN, DTRK, OUT_SOFTPLUS);

    scan_partial_kernel<<<dim3(BB * DIN / 256, NCHUNK), blk, 0, stream>>>(
        dtf, xcb, xdbl, A_log + (size_t)l * DIN * DSTATE, Aprod, Hpart);
    scan_carry_kernel<<<(BB * DIN * DSTATE + 255) / 256, blk, 0, stream>>>(Aprod, Hpart, Hinit);
    scan_final_kernel<<<dim3(BB * DIN / 256, NCHUNK), blk, 0, stream>>>(
        dtf, xcb, xdbl, xzb, A_log + (size_t)l * DIN * DSTATE,
        D_par + (size_t)l * DIN, Hinit, yb);

    wmma_gemm_kernel<<<dim3((DMODEL + BN - 1) / BN, MM / BM), blk, 0, stream>>>(
        yb, w_out, resid, (const float*)nullptr, MM, DMODEL, DIN, OUT_F32_ADD);
  }

  layernorm_kernel<<<MM, blk, 0, stream>>>(
      (const float*)nullptr, resid, fn_w, fn_b, (bf16*)nullptr, (float*)d_out);
}
